// StereographicAttention_24653112279108
// MI455X (gfx1250) — matrix operand order repaired and checked by reading, not compiled, not run on hardware
//
#include <hip/hip_runtime.h>
#include <math.h>

#define NN    16384
#define DIMK  1024
#define HDIM  1024
#define NH    16
#define HD    64
#define EPSF  1e-15f
#define MAXN  0.996f   // (1 - PROJ_EPS) / sqrt(|k|), k = -1
#define PITCH 144      // 128B row + 16B pad -> conflict-free LDS reads

typedef __bf16 bf16;
typedef bf16  v4bf  __attribute__((ext_vector_type(4)));
typedef bf16  v8bf  __attribute__((ext_vector_type(8)));
typedef bf16  v16bf __attribute__((ext_vector_type(16)));
typedef float v8f   __attribute__((ext_vector_type(8)));
typedef unsigned int u32x4 __attribute__((ext_vector_type(4)));
typedef int i32x8 __attribute__((ext_vector_type(8)));
typedef int i32x4 __attribute__((ext_vector_type(4)));

// ---------------------------------------------------------------- helpers
__device__ __forceinline__ float clip1(float x) {
  return fminf(fmaxf(x, -1.f + 1e-7f), 1.f - 1e-7f);
}
__device__ __forceinline__ float eluf(float z) { return z > 0.f ? z : expm1f(z); }

__device__ __forceinline__ float sum16(float v) {
  #pragma unroll
  for (int m = 8; m > 0; m >>= 1) v += __shfl_xor(v, m, 16);
  return v;
}

__device__ __forceinline__ float blockSum(float v, float* red) {
  #pragma unroll
  for (int m = 16; m > 0; m >>= 1) v += __shfl_xor(v, m, 32);
  const int wv = threadIdx.x >> 5, ln = threadIdx.x & 31;
  if (ln == 0) red[wv] = v;
  __syncthreads();
  float s = (threadIdx.x < 8) ? red[threadIdx.x] : 0.f;
  if (wv == 0) {
    #pragma unroll
    for (int m = 4; m > 0; m >>= 1) s += __shfl_xor(s, m, 32);
    if (ln == 0) red[0] = s;
  }
  __syncthreads();
  float r = red[0];
  __syncthreads();
  return r;
}

__device__ __forceinline__ v16bf combine16(v8bf lo, v8bf hi) {
  v16bf r;
  #pragma unroll
  for (int i = 0; i < 8; i++) { r[i] = lo[i]; r[8 + i] = hi[i]; }
  return r;
}

// CDNA5 async global->LDS copy (ASYNCcnt tracked)
__device__ __forceinline__ void async_ld_b128(unsigned lds_off, const void* gaddr) {
  asm volatile("global_load_async_to_lds_b128 %0, %1, off"
               :: "v"(lds_off), "v"(gaddr) : "memory");
}
__device__ __forceinline__ void wait_async_le2() {
  asm volatile("s_wait_asynccnt 0x2" ::: "memory");
}
__device__ __forceinline__ void wait_async_le0() {
  asm volatile("s_wait_asynccnt 0x0" ::: "memory");
}

// möbius-matvec tail + kappa_relu over a 1024-dim row (4 elems/thread, 256 thr)
__device__ __forceinline__ void mobius_kappa_row(float* mx, float xn, float* red) {
  float sm = mx[0]*mx[0] + mx[1]*mx[1] + mx[2]*mx[2] + mx[3]*mx[3];
  float smt = blockSum(sm, red);
  float mxn_raw = sqrtf(smt);
  float mxn = fmaxf(mxn_raw, EPSF);
  float c1 = (mxn_raw > 0.f)
           ? tanhf((mxn / xn) * atanhf(clip1(xn))) / mxn : 0.f;
  float v[4], sv = 0.f;
  #pragma unroll
  for (int i = 0; i < 4; i++) { v[i] = c1 * mx[i]; sv += v[i]*v[i]; }
  float yn = fmaxf(sqrtf(blockSum(sv, red)), EPSF);
  float c2 = atanhf(clip1(yn)) / yn;               // logmap0 scale
  float u[4], su = 0.f;
  #pragma unroll
  for (int i = 0; i < 4; i++) { u[i] = fmaxf(c2 * v[i], 0.f); su += u[i]*u[i]; }
  float sut = blockSum(su, red);
  float un = fmaxf(sqrtf(sut), EPSF);
  float c3 = tanhf(un) / un;                       // expmap0 scale
  float wn = c3 * sqrtf(sut);
  float sc = (wn > MAXN) ? (MAXN / wn) : 1.f;      // project
  #pragma unroll
  for (int i = 0; i < 4; i++) mx[i] = sc * c3 * u[i];
}

// ---------------------------------------------------------------- kernels

__global__ __launch_bounds__(256)
void transpose_bf16_kernel(const float* __restrict__ W, bf16* __restrict__ Wt,
                           int R, int C) {
  __shared__ float tile[32][33];
  const int bx = blockIdx.x * 32, by = blockIdx.y * 32;
  const int tx = threadIdx.x & 31, ty = threadIdx.x >> 5;
  #pragma unroll
  for (int i = 0; i < 32; i += 8)
    tile[ty + i][tx] = W[(size_t)(by + ty + i) * C + bx + tx];
  __syncthreads();
  #pragma unroll
  for (int i = 0; i < 32; i += 8)
    Wt[(size_t)(bx + ty + i) * R + by + tx] = (bf16)tile[tx][ty + i];
}

__global__ __launch_bounds__(256)
void convx_kernel(const float* __restrict__ X, bf16* __restrict__ Xb) {
  size_t i = ((size_t)blockIdx.x * 256 + threadIdx.x) * 4;
  float4 v = *(const float4*)(X + i);
  v4bf o = { (bf16)v.x, (bf16)v.y, (bf16)v.z, (bf16)v.w };
  *(v4bf*)(Xb + i) = o;
}

__global__ void zerof_kernel(float* p, int n) {
  int i = blockIdx.x * 256 + threadIdx.x;
  if (i < n) p[i] = 0.f;
}

// C[M,N] = A[M,K](bf16) * Bt[N,K]^T(bf16) (+bias).
// Block: 8 waves, 256x64 tile; wave = 32x64. K-step 64.
// B slab (64 N-rows x 64 K) double-buffered in LDS via async global->LDS.
template<bool OUT_BF16, bool HAS_BIAS>
__global__ __launch_bounds__(256)
void gemm_wmma_kernel(const bf16* __restrict__ A, const bf16* __restrict__ Bt,
                      const float* __restrict__ bias, void* __restrict__ Cout,
                      int M, int N, int K) {
  __shared__ __align__(16) unsigned char sB[2 * 64 * PITCH];
  const int t = threadIdx.x;
  const int lane = t & 31, wv = t >> 5;
  const int nb0 = blockIdx.x * 64;
  const int m0 = blockIdx.y * 256 + wv * 32;
  const int arow = m0 + (lane & 15);
  const int ka = (lane >> 4) * 8;     // A K-sub offset per ISA layout
  const int kb = (lane >> 4) * 8;     // B K-sub offset per ISA layout
  const bf16* Ap = A + (size_t)arow * K;

  // staging: 512 chunks of 16B (64 rows x 8 segs), 2 chunks/thread
  const int srow = t >> 3, sseg = t & 7;
  const unsigned sbase = (unsigned)(size_t)(&sB[0]);
  const unsigned soff = sbase + srow * PITCH + sseg * 16;
  const bf16* Bg = Bt + (size_t)(nb0 + srow) * K + sseg * 8;

  v8f acc[2][4];
  #pragma unroll
  for (int mt = 0; mt < 2; mt++)
    #pragma unroll
    for (int i = 0; i < 4; i++) { v8f z = {0,0,0,0,0,0,0,0}; acc[mt][i] = z; }

  // prologue: stage slab 0 into buffer 0
  async_ld_b128(soff, (const void*)(Bg));
  async_ld_b128(soff + 32 * PITCH, (const void*)(Bg + (size_t)32 * K));

  for (int k0 = 0; k0 < K; k0 += 64) {
    const int cur = (k0 >> 6) & 1;
    const bool has_next = (k0 + 64) < K;
    __syncthreads();                 // next buffer safe to overwrite
    if (has_next) {
      const unsigned nb = sbase + (cur ^ 1) * (64 * PITCH)
                        + srow * PITCH + sseg * 16;
      async_ld_b128(nb, (const void*)(Bg + k0 + 64));
      async_ld_b128(nb + 32 * PITCH, (const void*)(Bg + k0 + 64 + (size_t)32 * K));
    }
    // A fragments: 2 row-tiles x 2 k-steps
    v16bf a[2][2];
    #pragma unroll
    for (int mt = 0; mt < 2; mt++)
      #pragma unroll
      for (int kk = 0; kk < 2; kk++) {
        const bf16* ap = Ap + (size_t)mt * 16 * K + k0 + kk * 32 + ka;
        a[mt][kk] = combine16(*(const v8bf*)(ap), *(const v8bf*)(ap + 16));
      }
    if (has_next) wait_async_le2(); else wait_async_le0();
    __syncthreads();                 // current slab complete for all waves
    const unsigned char* buf = sB + cur * (64 * PITCH);
    #pragma unroll
    for (int kk = 0; kk < 2; kk++) {
      #pragma unroll
      for (int tn = 0; tn < 4; tn++) {
        const unsigned char* bp = buf + (tn * 16 + (lane & 15)) * PITCH
                                + (kk * 32 + kb) * 2;
        v16bf b = combine16(*(const v8bf*)(bp), *(const v8bf*)(bp + 32));
        #pragma unroll
        for (int mt = 0; mt < 2; mt++)
          acc[mt][tn] = __builtin_amdgcn_wmma_f32_16x16x32_bf16(
              false, a[mt][kk], false, b, (short)0, acc[mt][tn], false, false);
      }
    }
  }
  #pragma unroll
  for (int mt = 0; mt < 2; mt++)
    #pragma unroll
    for (int tn = 0; tn < 4; tn++) {
      const int col = nb0 + tn * 16 + (lane & 15);
      const float bv = HAS_BIAS ? bias[col] : 0.f;
      #pragma unroll
      for (int r = 0; r < 8; r++) {
        const int mrow = m0 + mt * 16 + r + (lane >> 4) * 8;
        const float v = acc[mt][tn][r] + bv;
        if (OUT_BF16) ((bf16*)Cout)[(size_t)mrow * N + col] = (bf16)v;
        else          ((float*)Cout)[(size_t)mrow * N + col] = v;
      }
    }
}

// V path: per row, möbius_matvec tail + kappa_relu, per-head denom & x matrix
__global__ __launch_bounds__(256)
void v_transform_kernel(const float* __restrict__ X, const float* __restrict__ Vraw,
                        const float* __restrict__ mask,
                        bf16* __restrict__ xb, float* __restrict__ denomv) {
  __shared__ float red[8];
  const int n = blockIdx.x, t = threadIdx.x, j0 = t * 4, h = t >> 4;
  float4 xr = *(const float4*)(X + (size_t)n * DIMK + j0);
  float sx = xr.x*xr.x + xr.y*xr.y + xr.z*xr.z + xr.w*xr.w;
  float xn = fmaxf(sqrtf(blockSum(sx, red)), EPSF);
  float4 m4 = *(const float4*)(Vraw + (size_t)n * HDIM + j0);
  float mx[4] = { m4.x, m4.y, m4.z, m4.w };
  mobius_kappa_row(mx, xn, red);           // mx = final V values
  float sv = mx[0]*mx[0] + mx[1]*mx[1] + mx[2]*mx[2] + mx[3]*mx[3];
  float hs = sum16(sv);                    // per-head |V|^2
  float dv = fmaxf(1.f - hs, EPSF);        // 1 + k*|V|^2, k=-1
  float gamma = 2.f / dv, gm1 = gamma - 1.f;
  float sgn = (gm1 >= 0.f) ? 1.f : -1.f;
  float denr = sgn * fmaxf(fabsf(gm1), 1e-10f);
  float xf = gamma / denr * mask[n];
  v4bf xo = { (bf16)(xf*mx[0]), (bf16)(xf*mx[1]), (bf16)(xf*mx[2]), (bf16)(xf*mx[3]) };
  *(v4bf*)(xb + (size_t)n * HDIM + j0) = xo;
  if ((t & 15) == 0) denomv[n * NH + h] = dv;
}

__global__ __launch_bounds__(256)
void qk_stats_kernel(const bf16* __restrict__ Qb, const bf16* __restrict__ Kb,
                     const float* __restrict__ denomv, const float* __restrict__ mask,
                     bf16* __restrict__ v1b, bf16* __restrict__ v2b) {
  const int n = blockIdx.x, t = threadIdx.x, j0 = t * 4, h = t >> 4;
  const float dv = denomv[n * NH + h];
  const float gamma = 2.f / dv, gm1 = gamma - 1.f;
  const float sgn = (gm1 >= 0.f) ? 1.f : -1.f;
  const float denr = sgn * fmaxf(fabsf(gm1), 1e-10f);
  const float m = mask[n];
  v4bf q4 = *(const v4bf*)(Qb + (size_t)n * HDIM + j0);
  v4bf k4 = *(const v4bf*)(Kb + (size_t)n * HDIM + j0);
  v4bf o1, o2;
  #pragma unroll
  for (int i = 0; i < 4; i++) {
    o1[i] = (bf16)(eluf((float)q4[i] / dv) + 1.f);
    o2[i] = (bf16)(denr * (eluf((float)k4[i] / dv) + 1.f) * m);
  }
  *(v4bf*)(v1b + (size_t)n * HDIM + j0) = o1;
  *(v4bf*)(v2b + (size_t)n * HDIM + j0) = o2;
}

__global__ __launch_bounds__(256)
void colsum_kernel(const bf16* __restrict__ v2b, float* __restrict__ v2sum) {
  __shared__ float part[256];
  const int c0 = blockIdx.x * 16, cr = threadIdx.x & 15, rr = threadIdx.x >> 4;
  float acc = 0.f;
  for (int nrow = rr; nrow < NN; nrow += 16)
    acc += (float)v2b[(size_t)nrow * HDIM + c0 + cr];
  part[threadIdx.x] = acc;
  __syncthreads();
  if (threadIdx.x < 16) {
    float s = 0.f;
    #pragma unroll
    for (int r = 0; r < 16; r++) s += part[r * 16 + threadIdx.x];
    v2sum[c0 + threadIdx.x] = s;
  }
}

__global__ __launch_bounds__(256)
void dcalc_kernel(const bf16* __restrict__ v1b, const float* __restrict__ v2sum,
                  float* __restrict__ dinv) {
  const int n = blockIdx.x, t = threadIdx.x, h = t >> 4, d0 = (t & 15) * 4;
  const bf16* p = v1b + (size_t)n * HDIM + h * HD + d0;
  const float* q = v2sum + h * HD + d0;
  float s = 0.f;
  #pragma unroll
  for (int i = 0; i < 4; i++) s += (float)p[i] * q[i];
  s = sum16(s);
  if ((t & 15) == 0) {
    float D = (s == 0.f) ? 1e-5f : s;
    dinv[n * NH + h] = 1.f / D;
  }
}

// context^T accumulation: cT[h][e][d] += sum_n v2[n,d]*x[n,e]
__global__ __launch_bounds__(256)
void context_kernel(const bf16* __restrict__ v2b, const bf16* __restrict__ xb,
                    float* __restrict__ cT) {
  __shared__ bf16 sv[8][64];
  __shared__ bf16 sx[8][64];
  const int h = blockIdx.y;
  const int nbase = blockIdx.x * (NN / 16);
  const int t = threadIdx.x;
  const int d = t >> 2, e0 = (t & 3) * 16;
  float acc[16];
  #pragma unroll
  for (int i = 0; i < 16; i++) acc[i] = 0.f;
  for (int nb = 0; nb < NN / 16; nb += 8) {
    __syncthreads();
    const int id = t * 4, nn = id >> 7, w = id & 127;
    const size_t base = (size_t)(nbase + nb + nn) * HDIM + h * HD;
    if (w < 64) *(v4bf*)&sv[nn][w]      = *(const v4bf*)(v2b + base + w);
    else        *(v4bf*)&sx[nn][w - 64] = *(const v4bf*)(xb  + base + (w - 64));
    __syncthreads();
    #pragma unroll
    for (int k = 0; k < 8; k++) {
      const float v2d = (float)sv[k][d];
      #pragma unroll
      for (int i = 0; i < 16; i++) acc[i] += v2d * (float)sx[k][e0 + i];
    }
  }
  #pragma unroll
  for (int i = 0; i < 16; i++)
    atomicAdd(&cT[((h * HD) + e0 + i) * HD + d], acc[i]);
}

__global__ void ct_conv_kernel(const float* __restrict__ cT, bf16* __restrict__ cTb) {
  int i = blockIdx.x * 256 + threadIdx.x;
  cTb[i] = (bf16)cT[i];
}

// attn[node, h*64+e] = (sum_d v1[node,h,d] * cT[h][e][d]) * dinv[node,h]
// Per-head 64x64 context^T staged into LDS by the Tensor Data Mover with
// row padding (128B rows -> 144B pitch) for conflict-free B-fragment reads.
__global__ __launch_bounds__(256)
void attn_out_wmma_kernel(const bf16* __restrict__ v1b, const bf16* __restrict__ cTb,
                          const float* __restrict__ dinv, float* __restrict__ attn) {
  __shared__ __align__(16) unsigned char sCt[64 * PITCH];
  const int t = threadIdx.x, lane = t & 31, wv = t >> 5;
  const int h = blockIdx.x;
  const int m0 = blockIdx.y * 128 + wv * 16;
  const int arow = m0 + (lane & 15);
  const int ka = (lane >> 4) * 8, kb = (lane >> 4) * 8;
  const bf16* Ap = v1b + (size_t)arow * HDIM + h * HD;

  if (wv == 0) {
    const unsigned ldsoff = (unsigned)(size_t)(&sCt[0]);
    const unsigned long long ga =
        (unsigned long long)(size_t)(cTb + (size_t)h * HD * HD);
    // D# group 0: count=1, lds_addr, global_addr[56:0], type=2
    u32x4 g0 = { 1u, ldsoff, (unsigned)ga,
                 (unsigned)((ga >> 32) & 0x01FFFFFFu) | (2u << 30) };
    // D# group 1: data_size=2B, pad_enable, pad_interval=32 DW (128B),
    // pad_amount=4 DW (16B); tensor 64x64, tile 64x64, dim0 stride 64
    i32x8 g1 = { (1 << 16) | (1 << 20) | (4 << 22) | (3 << 25),
                 (HD << 16),            // tensor_dim0 = 64
                 (HD << 16),            // tensor_dim1 = 64 (low16 of dim0 hi = 0)
                 (HD << 16),            // tile_dim0 = 64 (low16 of dim1 hi = 0)
                 HD,                    // tile_dim1 = 64, tile_dim2 = 0
                 HD,                    // tensor_dim0_stride = 64
                 0, 0 };
    i32x4 z4 = { 0, 0, 0, 0 };
    i32x8 z8 = { 0, 0, 0, 0, 0, 0, 0, 0 };
    __builtin_amdgcn_tensor_load_to_lds(g0, g1, z4, z4, z8, 0);
    __builtin_amdgcn_s_wait_tensorcnt(0);
  }
  __syncthreads();

  v8f acc[4];
  #pragma unroll
  for (int i = 0; i < 4; i++) { v8f z = {0,0,0,0,0,0,0,0}; acc[i] = z; }
  #pragma unroll
  for (int k0 = 0; k0 < HD; k0 += 32) {
    v16bf a = combine16(*(const v8bf*)(Ap + k0 + ka),
                        *(const v8bf*)(Ap + k0 + ka + 16));
    #pragma unroll
    for (int tn = 0; tn < 4; tn++) {
      const unsigned char* bp = sCt + (tn * 16 + (lane & 15)) * PITCH
                              + (k0 + kb) * 2;
      v16bf b = combine16(*(const v8bf*)(bp), *(const v8bf*)(bp + 32));
      acc[tn] = __builtin_amdgcn_wmma_f32_16x16x32_bf16(
          false, a, false, b, (short)0, acc[tn], false, false);
    }
  }
  #pragma unroll
  for (int tn = 0; tn < 4; tn++) {
    const int e = tn * 16 + (lane & 15);
    #pragma unroll
    for (int r = 0; r < 8; r++) {
      const int node = m0 + r + (lane >> 4) * 8;
      attn[(size_t)node * HDIM + h * HD + e] = acc[tn][r] * dinv[node * NH + h];
    }
  }
}

// per (node, head): project -> mobius_scalar_mul(0.5) -> project; build comb row
__global__ __launch_bounds__(256)
void attn_epi_kernel(const float* __restrict__ attn, bf16* __restrict__ comb,
                     float* __restrict__ combnorm) {
  __shared__ float red[8];
  const int n = blockIdx.x, t = threadIdx.x, j0 = t * 4;
  float4 o4 = *(const float4*)(attn + (size_t)n * HDIM + j0);
  float o[4] = { o4.x, o4.y, o4.z, o4.w };
  float ss = o[0]*o[0] + o[1]*o[1] + o[2]*o[2] + o[3]*o[3];
  float hn = sum16(ss);                            // per-head |o|^2
  float n1 = fmaxf(sqrtf(hn), EPSF);
  float sc1 = (n1 > MAXN) ? (MAXN / n1) : 1.f;     // project
  float on = fminf(n1, MAXN);
  float c = tanhf(0.5f * atanhf(clip1(on))) / on;  // mobius scalar mul 0.5
  float nn2 = c * on;
  float sc2 = (nn2 > MAXN) ? (MAXN / nn2) : 1.f;   // project
  float f = sc1 * c * sc2;
  float fin[4], s2 = 0.f;
  #pragma unroll
  for (int i = 0; i < 4; i++) { fin[i] = f * o[i]; s2 += fin[i]*fin[i]; }
  v4bf co = { (bf16)fin[0], (bf16)fin[1], (bf16)fin[2], (bf16)fin[3] };
  *(v4bf*)(comb + (size_t)n * HDIM + j0) = co;
  float tot = blockSum(s2, red);
  if (t == 0) combnorm[n] = fmaxf(sqrtf(tot), EPSF);
}

__global__ __launch_bounds__(256)
void final_rows_kernel(const float* __restrict__ mxff, const float* __restrict__ combnorm,
                       float* __restrict__ out) {
  __shared__ float red[8];
  const int n = blockIdx.x, t = threadIdx.x, j0 = t * 4;
  float4 m4 = *(const float4*)(mxff + (size_t)n * DIMK + j0);
  float mx[4] = { m4.x, m4.y, m4.z, m4.w };
  mobius_kappa_row(mx, combnorm[n], red);
  float4 o = { mx[0], mx[1], mx[2], mx[3] };
  *(float4*)(out + (size_t)n * DIMK + j0) = o;
}

// ---------------------------------------------------------------- launch
extern "C" void kernel_launch(void* const* d_in, const int* in_sizes, int n_in,
                              void* d_out, int out_size, void* d_ws, size_t ws_size,
                              hipStream_t stream) {
  (void)in_sizes; (void)n_in; (void)out_size; (void)ws_size;
  const float* X    = (const float*)d_in[0];
  const float* mask = (const float*)d_in[1];
  const float* Wq   = (const float*)d_in[2];
  const float* bq   = (const float*)d_in[3];
  const float* Wk   = (const float*)d_in[4];
  const float* bk   = (const float*)d_in[5];
  const float* Wv   = (const float*)d_in[6];
  const float* Wff  = (const float*)d_in[7];
  float* out = (float*)d_out;
  char*  ws  = (char*)d_ws;
  const size_t MB = 1ull << 20;

  bf16*  Xb   = (bf16*)(ws);               // 32 MB
  bf16*  WqT  = (bf16*)(ws + 32  * MB);    // 2 MB each
  bf16*  WkT  = (bf16*)(ws + 34  * MB);
  bf16*  WvT  = (bf16*)(ws + 36  * MB);
  bf16*  WffT = (bf16*)(ws + 38  * MB);
  bf16*  Qb   = (bf16*)(ws + 40  * MB);    // 32 MB
  bf16*  Kb   = (bf16*)(ws + 72  * MB);    // 32 MB
  bf16*  v1b  = (bf16*)(ws + 104 * MB);    // 32 MB
  bf16*  v2b  = (bf16*)(ws + 136 * MB);    // 32 MB
  bf16*  xb   = (bf16*)(ws + 168 * MB);    // 32 MB
  bf16*  comb = (bf16*)(ws + 200 * MB);    // 32 MB
  float* big  = (float*)(ws + 232 * MB);   // 64 MB: Vraw -> attn_raw -> mxff
  float* denomv   = (float*)(ws + 296 * MB);
  float* dinv     = (float*)(ws + 297 * MB);
  float* v2sum    = (float*)(ws + 298 * MB);
  float* cT       = (float*)(ws + 298 * MB + (64 << 10));
  bf16*  cTb      = (bf16*) (ws + 298 * MB + (64 << 10) + (256 << 10));
  float* combnorm = (float*)(ws + 298 * MB + (64 << 10) + (256 << 10) + (128 << 10));

  dim3 b256(256);
  transpose_bf16_kernel<<<dim3(32, 32), b256, 0, stream>>>(Wq,  WqT,  DIMK, HDIM);
  transpose_bf16_kernel<<<dim3(32, 32), b256, 0, stream>>>(Wk,  WkT,  DIMK, HDIM);
  transpose_bf16_kernel<<<dim3(32, 32), b256, 0, stream>>>(Wv,  WvT,  DIMK, HDIM);
  transpose_bf16_kernel<<<dim3(32, 32), b256, 0, stream>>>(Wff, WffT, HDIM, DIMK);
  convx_kernel<<<NN * DIMK / 1024, b256, 0, stream>>>(X, Xb);

  dim3 gg(HDIM / 64, NN / 256);
  gemm_wmma_kernel<true,  true ><<<gg, b256, 0, stream>>>(Xb, WqT, bq, Qb, NN, HDIM, DIMK);
  gemm_wmma_kernel<true,  true ><<<gg, b256, 0, stream>>>(Xb, WkT, bk, Kb, NN, HDIM, DIMK);
  gemm_wmma_kernel<false, false><<<gg, b256, 0, stream>>>(Xb, WvT, nullptr, big, NN, HDIM, DIMK);

  v_transform_kernel<<<NN, b256, 0, stream>>>(X, big, mask, xb, denomv);
  qk_stats_kernel<<<NN, b256, 0, stream>>>(Qb, Kb, denomv, mask, v1b, v2b);
  colsum_kernel<<<HDIM / 16, b256, 0, stream>>>(v2b, v2sum);
  dcalc_kernel<<<NN, b256, 0, stream>>>(v1b, v2sum, dinv);

  zerof_kernel<<<(NH * HD * HD + 255) / 256, b256, 0, stream>>>(cT, NH * HD * HD);
  context_kernel<<<dim3(16, NH), b256, 0, stream>>>(v2b, xb, cT);
  ct_conv_kernel<<<NH * HD * HD / 256, b256, 0, stream>>>(cT, cTb);

  attn_out_wmma_kernel<<<dim3(NH, NN / 128), b256, 0, stream>>>(v1b, cTb, dinv, big);
  attn_epi_kernel<<<NN, b256, 0, stream>>>(big, comb, combnorm);

  gemm_wmma_kernel<false, false><<<dim3(DIMK / 64, NN / 256), b256, 0, stream>>>(
      comb, WffT, nullptr, big, NN, DIMK, HDIM);
  final_rows_kernel<<<NN, b256, 0, stream>>>(big, combnorm, out);
}
